// AffordancePredictor_43894565765408
// MI455X (gfx1250) — compile-verified, matrix operand order checked by reading, not run on hardware
//
#include <hip/hip_runtime.h>
#include <hip/hip_bf16.h>
#include <cstdint>

typedef _Float16 h16;
typedef __attribute__((ext_vector_type(16))) _Float16 v16h;
typedef __attribute__((ext_vector_type(8)))  _Float16 h8;
typedef __attribute__((ext_vector_type(8)))  float    v8f;

#if defined(__has_builtin)
#if __has_builtin(__builtin_amdgcn_global_load_async_to_lds_b128) && \
    __has_builtin(__builtin_amdgcn_s_wait_asynccnt)
#define USE_ASYNC_LDS 1
#endif
#endif
#ifndef USE_ASYNC_LDS
#define USE_ASYNC_LDS 0
#endif

#if USE_ASYNC_LDS
typedef int i4v __attribute__((vector_size(16)));
// GLOBAL_LOAD_ASYNC_TO_LDS_B128: per-lane 16B global -> LDS copy (ASYNCcnt).
__device__ __forceinline__ void async_copy16(const h16* g, h16* l) {
  __builtin_amdgcn_global_load_async_to_lds_b128(
      (__attribute__((address_space(1))) i4v*)(void*)const_cast<h16*>(g),
      (__attribute__((address_space(3))) i4v*)(void*)l, 0, 0);
}
#endif

// K index held by fragment element e (0..15) for a lane in half = lane>>4,
// per CDNA5 ISA 16-bit A-matrix 16x32 layout (B symmetric, N per lane&15):
//   e < 8 : K = half*8 + e ;  e >= 8 : K = 16 + half*8 + (e-8)
__device__ __forceinline__ int wmma_kmap(int e, int half) {
  return (e < 8) ? (half * 8 + e) : (16 + half * 8 + (e - 8));
}

// ---------------- weight staging ----------------------------------------

__global__ void bn_fold(const float* __restrict__ g, const float* __restrict__ b,
                        const float* __restrict__ m, const float* __restrict__ v,
                        float* __restrict__ scale, float* __restrict__ bias, int C) {
  int i = blockIdx.x * blockDim.x + threadIdx.x;
  if (i >= C) return;
  float s = g[i] * rsqrtf(v[i] + 1e-5f);
  scale[i] = s;
  bias[i]  = b[i] - m[i] * s;
}

// OIHW f32 -> fragment-ordered f16:  [(ntile*kt + ktile)*32 + lane]*16 + e
__global__ void repack_w(const float* __restrict__ w, h16* __restrict__ out,
                         int Cout, int Cin, int KHW) {
  int K = KHW * KHW * Cin;
  int kt = K >> 5;
  int idx = blockIdx.x * blockDim.x + threadIdx.x;
  int total = kt * (Cout >> 4) * 512;
  if (idx >= total) return;
  int e    = idx & 15;
  int lane = (idx >> 4) & 31;
  int t    = idx >> 9;
  int kti  = t % kt, nti = t / kt;
  int n    = (nti << 4) + (lane & 15);
  int half = lane >> 4;
  int k    = (kti << 5) + wmma_kmap(e, half);
  int kh = k / (KHW * Cin);
  int r  = k - kh * (KHW * Cin);
  int kw = r / Cin;
  int ci = r - kw * Cin;
  out[idx] = (h16)w[(((size_t)n * Cin + ci) * KHW + kh) * KHW + kw];
}

// ---------------- conv1 (7x7 s2 p3) direct, NCHW f32 -> NHWC f16 --------

__global__ void conv1_bn_relu(const float* __restrict__ img, const float* __restrict__ w,
                              const float* __restrict__ scale, const float* __restrict__ bias,
                              h16* __restrict__ out) {
  size_t idx = (size_t)blockIdx.x * blockDim.x + threadIdx.x;
  const size_t total = (size_t)64 * 112 * 112 * 64;
  if (idx >= total) return;
  int co = idx & 63;
  size_t pos = idx >> 6;
  int wo = pos % 112; size_t t = pos / 112;
  int ho = t % 112;   int n = (int)(t / 112);
  float acc = 0.f;
  for (int ci = 0; ci < 3; ++ci)
    for (int kh = 0; kh < 7; ++kh) {
      int ih = ho * 2 - 3 + kh;
      if (ih < 0 || ih >= 224) continue;
      for (int kw = 0; kw < 7; ++kw) {
        int iw = wo * 2 - 3 + kw;
        if (iw < 0 || iw >= 224) continue;
        acc += img[(((size_t)n * 3 + ci) * 224 + ih) * 224 + iw] *
               w[(((size_t)co * 3 + ci) * 7 + kh) * 7 + kw];
      }
    }
  acc = acc * scale[co] + bias[co];
  out[idx] = (h16)fmaxf(acc, 0.f);
}

// ---------------- maxpool 3x3 s2 p1, NHWC f16, 112 -> 56, C=64 ----------

__global__ void maxpool_k(const h16* __restrict__ in, h16* __restrict__ out) {
  size_t idx = (size_t)blockIdx.x * blockDim.x + threadIdx.x;
  const size_t total = (size_t)64 * 56 * 56 * 64;
  if (idx >= total) return;
  int c = idx & 63;
  size_t pos = idx >> 6;
  int wo = pos % 56; size_t t = pos / 56;
  int ho = t % 56;   int n = (int)(t / 56);
  float m = -3.4e38f;
  for (int kh = 0; kh < 3; ++kh) {
    int ih = ho * 2 - 1 + kh;
    if (ih < 0 || ih >= 112) continue;
    for (int kw = 0; kw < 3; ++kw) {
      int iw = wo * 2 - 1 + kw;
      if (iw < 0 || iw >= 112) continue;
      m = fmaxf(m, (float)in[(((size_t)n * 112 + ih) * 112 + iw) * 64 + c]);
    }
  }
  out[idx] = (h16)m;
}

// ---------------- WMMA implicit-GEMM conv + BN (+res)(+relu) ------------
// x: NHWC f16; wf: fragment-packed f16; out/res: NHWC f16 (M = N*Ho*Wo rows)
// block = 128 thr = 4 waves; block tile = 64(M) x 64(N); K chunk = 32.
// Double-buffered LDS A stage in fragment-major layout As[buf][row][half][e]:
// each lane's A fragment is one contiguous 32B LDS read; each loader thread
// stages two contiguous 16B global segments -> two 16B LDS slots via per-lane
// GLOBAL_LOAD_ASYNC_TO_LDS_B128 (ASYNCcnt); padding lanes ds-store zeros.
// Loader uses a (kh,kw,ci) cursor so the K loop contains no integer division.

template <bool RELU, bool ADDRES>
__global__ void conv_wmma(const h16* __restrict__ x, const h16* __restrict__ wf,
                          const float* __restrict__ scale, const float* __restrict__ bias,
                          const h16* __restrict__ res, h16* __restrict__ out,
                          int H, int W, int Cin, int Cout, int KHW, int stride, int pad,
                          int Ho, int Wo) {
  __shared__ __align__(32) h16 As[2][64][2][16];
  const int kt  = (KHW * KHW * Cin) >> 5;
  const int cpk = Cin >> 5;                 // K chunks per (kh,kw)
  const int tid  = threadIdx.x;
  const int lane = tid & 31;
  const int wave = tid >> 5;
  const int mtile = blockIdx.x;             // 64-row tile
  const int ntile = blockIdx.y * 4 + wave;  // 16-col tile
  const int ncol  = (ntile << 4) + (lane & 15);
  const int half  = lane >> 4;
  const int am    = lane & 15;

  // loader role: 2 threads per A row (each stages 16 halfs = one K half-chunk)
  const int rm  = tid >> 1;
  const int seg = tid & 1;
  int gm = (mtile << 6) + rm;
  int wo = gm % Wo; int tq = gm / Wo;
  int ho = tq % Ho; int n = tq / Ho;
  const int hbase = ho * stride - pad;
  const int wbase = wo * stride - pad;

  int lkh = 0, lkw = 0, lci = 0;            // loader cursor = next chunk to stage
  auto load_chunk = [&](int buf) {
    int ih = hbase + lkh, iw = wbase + lkw;
    bool valid = (ih >= 0 && ih < H && iw >= 0 && iw < W);
    const h16* sp = x + (((size_t)n * H + ih) * W + iw) * Cin + (lci << 5) + (seg << 4);
    // seg0: kl 0..7 -> (half0,e0..7), kl 8..15 -> (half1,e0..7)
    // seg1: kl16..23 -> (half0,e8..15), kl24..31 -> (half1,e8..15)
    h16* d0 = &As[buf][rm][0][seg << 3];
    h16* d1 = &As[buf][rm][1][seg << 3];
#if USE_ASYNC_LDS
    if (valid) {
      async_copy16(sp, d0);
      async_copy16(sp + 8, d1);
    } else {
      *(h8*)d0 = (h8){};
      *(h8*)d1 = (h8){};
    }
#else
    h8 v0 = {}, v1 = {};
    if (valid) {
      v0 = *(const h8*)sp;
      v1 = *(const h8*)(sp + 8);
    }
    *(h8*)d0 = v0;
    *(h8*)d1 = v1;
#endif
    if (++lci == cpk) { lci = 0; if (++lkw == KHW) { lkw = 0; ++lkh; } }
  };

  v8f acc[4] = {{}, {}, {}, {}};
  load_chunk(0);
#if USE_ASYNC_LDS
  __builtin_amdgcn_s_wait_asynccnt(0);
#endif
  __syncthreads();
  for (int kc = 0; kc < kt; ++kc) {
    int cur = kc & 1;
    if (kc + 1 < kt) load_chunk(cur ^ 1);   // stage next chunk into other buffer
    const h16* bp = wf + (((size_t)ntile * kt + kc) * 32 + lane) * 16;
    v16h bv = *(const v16h*)bp;
    if (kc + 1 < kt)
      __builtin_prefetch(wf + (((size_t)ntile * kt + kc + 1) * 32 + lane) * 16, 0, 1);
#pragma unroll
    for (int mt = 0; mt < 4; ++mt) {
      v16h av;
#pragma unroll
      for (int e = 0; e < 16; ++e) av[e] = As[cur][(mt << 4) + am][half][e];
      acc[mt] = __builtin_amdgcn_wmma_f32_16x16x32_f16(false, av, false, bv, (short)0,
                                                       acc[mt], false, false);
    }
#if USE_ASYNC_LDS
    __builtin_amdgcn_s_wait_asynccnt(0);    // drain copies for the other buffer
#endif
    __syncthreads();
  }

  float sc = scale[ncol], bi = bias[ncol];
#pragma unroll
  for (int mt = 0; mt < 4; ++mt) {
#pragma unroll
    for (int r = 0; r < 8; ++r) {
      int m = (mtile << 6) + (mt << 4) + (half << 3) + r;  // C/D: M = half*8+r
      size_t oi = (size_t)m * Cout + ncol;
      float vv = acc[mt][r] * sc + bi;
      if (ADDRES) vv += (float)res[oi];
      if (RELU)   vv = fmaxf(vv, 0.f);
      out[oi] = (h16)vv;
    }
  }
}

// ---------------- heads --------------------------------------------------

__global__ void avgpool_k(const h16* __restrict__ in, float* __restrict__ out) {
  int idx = blockIdx.x * blockDim.x + threadIdx.x;
  if (idx >= 64 * 512) return;
  int c = idx & 511, n = idx >> 9;
  float s = 0.f;
  for (int p = 0; p < 49; ++p) s += (float)in[((size_t)n * 49 + p) * 512 + c];
  out[idx] = s * (1.f / 49.f);
}

// act: 0 none, 1 relu, 2 sigmoid.  W row-major [D,J]
__global__ void dense_k(const float* __restrict__ in, const float* __restrict__ W,
                        const float* __restrict__ bias, float* __restrict__ out,
                        int Bn, int D, int J, int act) {
  int idx = blockIdx.x * blockDim.x + threadIdx.x;
  if (idx >= Bn * J) return;
  int j = idx % J, b = idx / J;
  float acc = bias[j];
  const float* ip = in + (size_t)b * D;
  for (int d = 0; d < D; ++d) acc += ip[d] * W[(size_t)d * J + j];
  if (act == 1) acc = fmaxf(acc, 0.f);
  else if (act == 2) acc = 1.f / (1.f + __expf(-acc));
  out[idx] = acc;
}

// per-sample branch select: W [4,D,J], bias [4,J]
__global__ void bdense_k(const float* __restrict__ in, const float* __restrict__ W,
                         const float* __restrict__ bias, const int* __restrict__ cmd,
                         float* __restrict__ out, int Bn, int D, int J, int act) {
  int idx = blockIdx.x * blockDim.x + threadIdx.x;
  if (idx >= Bn * J) return;
  int j = idx % J, b = idx / J;
  int c = cmd[b];
  const float* Wb = W + (size_t)c * D * J;
  float acc = bias[(size_t)c * J + j];
  const float* ip = in + (size_t)b * D;
  for (int d = 0; d < D; ++d) acc += ip[d] * Wb[(size_t)d * J + j];
  if (act == 1) acc = fmaxf(acc, 0.f);
  out[idx] = acc;
}

// ---------------- host orchestration ------------------------------------
// Leaf order (JAX tree-flatten, dict keys sorted):
//  0 command, 1 img,
//  2..7   aff1 {b1,b2,b3,w1,w2,w3}, 8..13 aff2, 14..19 aff3, 20..25 aff4,
//  26..29 bn1 {b,g,m,v}, 30 conv1_w, 31 fc_b, 32 fc_w,
//  33..   layers: per block {bn1 b,g,m,v | bn2 b,g,m,v | [down_bn b,g,m,v | down_w] | w1 | w2}

extern "C" void kernel_launch(void* const* d_in, const int* in_sizes, int n_in,
                              void* d_out, int out_size, void* d_ws, size_t ws_size,
                              hipStream_t stream) {
  (void)in_sizes; (void)n_in; (void)out_size; (void)ws_size;
  const int*   cmd = (const int*)d_in[0];
  const float* img = (const float*)d_in[1];
  const int B = 64;

  char* wsb = (char*)d_ws;
  size_t off = 0;
  auto alloc = [&](size_t bytes) -> void* {
    off = (off + 255) & ~(size_t)255;
    void* p = wsb + off;
    off += bytes;
    return p;
  };

  auto fold = [&](int bidx, int C, float** sc, float** bi) {
    *sc = (float*)alloc(C * sizeof(float));
    *bi = (float*)alloc(C * sizeof(float));
    bn_fold<<<(C + 255) / 256, 256, 0, stream>>>(
        (const float*)d_in[bidx + 1], (const float*)d_in[bidx + 0],
        (const float*)d_in[bidx + 2], (const float*)d_in[bidx + 3], *sc, *bi, C);
  };
  auto repack = [&](int widx, int Cout, int Cin, int KHW) -> h16* {
    int K = KHW * KHW * Cin;
    h16* p = (h16*)alloc((size_t)K * Cout * sizeof(h16));
    int total = K * Cout;
    repack_w<<<(total + 255) / 256, 256, 0, stream>>>(
        (const float*)d_in[widx], p, Cout, Cin, KHW);
    return p;
  };

  // conv1 + maxpool
  float *sc1, *bi1;
  fold(26, 64, &sc1, &bi1);
  h16* act0 = (h16*)alloc((size_t)64 * 112 * 112 * 64 * sizeof(h16));
  {
    size_t tot = (size_t)64 * 112 * 112 * 64;
    conv1_bn_relu<<<(unsigned)((tot + 255) / 256), 256, 0, stream>>>(
        img, (const float*)d_in[30], sc1, bi1, act0);
  }
  h16* bufs[4];
  for (int i = 0; i < 4; ++i)
    bufs[i] = (h16*)alloc((size_t)64 * 56 * 56 * 64 * sizeof(h16));
  {
    size_t tot = (size_t)64 * 56 * 56 * 64;
    maxpool_k<<<(unsigned)((tot + 255) / 256), 256, 0, stream>>>(act0, bufs[0]);
  }

  auto launch_conv = [&](const h16* x, h16* wf, float* sc, float* bi, const h16* res,
                         h16* out, int H, int Cin, int Cout, int KHW, int stride,
                         int pad, int relu, int addres) {
    int Ho = (H + 2 * pad - KHW) / stride + 1;
    int M  = B * Ho * Ho;                 // divisible by 64 at every layer
    dim3 grid(M / 64, Cout / 64);
    if (relu && addres)
      conv_wmma<true, true><<<grid, 128, 0, stream>>>(x, wf, sc, bi, res, out, H, H,
                                                      Cin, Cout, KHW, stride, pad, Ho, Ho);
    else if (relu)
      conv_wmma<true, false><<<grid, 128, 0, stream>>>(x, wf, sc, bi, res, out, H, H,
                                                       Cin, Cout, KHW, stride, pad, Ho, Ho);
    else
      conv_wmma<false, false><<<grid, 128, 0, stream>>>(x, wf, sc, bi, res, out, H, H,
                                                        Cin, Cout, KHW, stride, pad, Ho, Ho);
  };

  struct BSpec { int w1, w2, dw, bn1, bn2, dbn, Cin, Cout, s, H; };
  const BSpec blocks[8] = {
      { 41,  42,  -1,  33,  37,  -1,  64,  64, 1, 56},
      { 51,  52,  -1,  43,  47,  -1,  64,  64, 1, 56},
      { 66,  67,  65,  53,  57,  61,  64, 128, 2, 56},
      { 76,  77,  -1,  68,  72,  -1, 128, 128, 1, 28},
      { 91,  92,  90,  78,  82,  86, 128, 256, 2, 28},
      {101, 102,  -1,  93,  97,  -1, 256, 256, 1, 14},
      {116, 117, 115, 103, 107, 111, 256, 512, 2, 14},
      {126, 127,  -1, 118, 122,  -1, 512, 512, 1,  7},
  };

  int xi = 0;
  for (int bI = 0; bI < 8; ++bI) {
    const BSpec& s = blocks[bI];
    float *s1, *b1, *s2, *b2, *sd = nullptr, *bd = nullptr;
    fold(s.bn1, s.Cout, &s1, &b1);
    fold(s.bn2, s.Cout, &s2, &b2);
    h16* w1 = repack(s.w1, s.Cout, s.Cin, 3);
    h16* w2 = repack(s.w2, s.Cout, s.Cout, 3);
    h16* wd = nullptr;
    if (s.dw >= 0) {
      fold(s.dbn, s.Cout, &sd, &bd);
      wd = repack(s.dw, s.Cout, s.Cin, 1);
    }
    int fr[3], fc2 = 0;
    for (int i = 0; i < 4; ++i)
      if (i != xi) fr[fc2++] = i;
    int ti = fr[0], ri = fr[1], oi = fr[2];

    launch_conv(bufs[xi], w1, s1, b1, nullptr, bufs[ti], s.H, s.Cin, s.Cout, 3, s.s, 1, 1, 0);
    const h16* res;
    if (wd) {
      launch_conv(bufs[xi], wd, sd, bd, nullptr, bufs[ri], s.H, s.Cin, s.Cout, 1, s.s, 0, 0, 0);
      res = bufs[ri];
    } else {
      res = bufs[xi];
    }
    int Hmid = (s.H + 2 - 3) / s.s + 1;
    launch_conv(bufs[ti], w2, s2, b2, res, bufs[oi], Hmid, s.Cout, s.Cout, 3, 1, 1, 1, 1);
    xi = oi;
  }

  // heads
  float* fbuf = (float*)alloc(B * 512 * sizeof(float));
  avgpool_k<<<(B * 512 + 255) / 256, 256, 0, stream>>>(bufs[xi], fbuf);
  float* fc = (float*)alloc(B * 512 * sizeof(float));
  dense_k<<<(B * 512 + 255) / 256, 256, 0, stream>>>(
      fbuf, (const float*)d_in[32], (const float*)d_in[31], fc, B, 512, 512, 0);

  float* h1 = (float*)alloc(B * 256 * sizeof(float));
  float* h2 = (float*)alloc(B * 256 * sizeof(float));
  float* outp = (float*)d_out;

  // aff1: 512->256->256->1, sigmoid
  dense_k<<<(B * 256 + 255) / 256, 256, 0, stream>>>(
      fc, (const float*)d_in[5], (const float*)d_in[2], h1, B, 512, 256, 1);
  dense_k<<<(B * 256 + 255) / 256, 256, 0, stream>>>(
      h1, (const float*)d_in[6], (const float*)d_in[3], h2, B, 256, 256, 1);
  dense_k<<<1, 64, 0, stream>>>(
      h2, (const float*)d_in[7], (const float*)d_in[4], outp, B, 256, 1, 2);

  // aff2..aff4: branched by command
  const int hb[3] = {8, 14, 20};
  for (int hI = 0; hI < 3; ++hI) {
    int bs = hb[hI];
    bdense_k<<<(B * 256 + 255) / 256, 256, 0, stream>>>(
        fc, (const float*)d_in[bs + 3], (const float*)d_in[bs + 0], cmd, h1, B, 512, 256, 1);
    bdense_k<<<(B * 256 + 255) / 256, 256, 0, stream>>>(
        h1, (const float*)d_in[bs + 4], (const float*)d_in[bs + 1], cmd, h2, B, 256, 256, 1);
    bdense_k<<<1, 64, 0, stream>>>(
        h2, (const float*)d_in[bs + 5], (const float*)d_in[bs + 2], cmd,
        outp + 64 * (hI + 1), B, 256, 1, 0);
  }
}
